// BipartiteEncoder_21431886807835
// MI455X (gfx1250) — matrix operand order repaired and checked by reading, not compiled, not run on hardware
//
#include <hip/hip_runtime.h>
#include <hip/hip_bf16.h>

typedef __attribute__((ext_vector_type(16))) _Float16 v16h;
typedef __attribute__((ext_vector_type(8)))  float    v8f;

#define N_NODES_C 100000
#define D_C 64
#define NTILES (N_NODES_C / 16)   // 6250 exact

// ---------------- utility kernels ----------------

__global__ void gcn_zero_f32(float* p, int n) {
    int i = blockIdx.x * blockDim.x + threadIdx.x;
    if (i < n) p[i] = 0.0f;
}

__global__ void gcn_degree(const int* __restrict__ dst, float* __restrict__ deg, int nE) {
    int i = blockIdx.x * blockDim.x + threadIdx.x;
    if (i < nE) unsafeAtomicAdd(&deg[dst[i]], 1.0f);
}

__global__ void gcn_dinv(const float* __restrict__ deg, float* __restrict__ dinv, int n) {
    int i = blockIdx.x * blockDim.x + threadIdx.x;
    if (i < n) dinv[i] = rsqrtf(deg[i] + 1.0f);
}

// Convert W1/W2 (64x64 f32, row-major K x N) to f16 stored column-major:
// WhT[n*64 + k] = (f16) W[k*64 + n]  -> B-fragment loads become contiguous.
__global__ void gcn_convW(const float* __restrict__ W1, const float* __restrict__ W2,
                          _Float16* __restrict__ Wh1, _Float16* __restrict__ Wh2) {
    int i = blockIdx.x * blockDim.x + threadIdx.x;   // 0..8191
    if (i >= 2 * D_C * D_C) return;
    const float* W = (i < D_C * D_C) ? W1 : W2;
    _Float16*   Wh = (i < D_C * D_C) ? Wh1 : Wh2;
    int j = i & (D_C * D_C - 1);
    int k = j >> 6, n = j & 63;
    Wh[n * D_C + k] = (_Float16)W[k * D_C + n];
}

// ---------------- WMMA GEMM: g = s = (X @ W) * dinv[m]  ----------------
// One wave per 16-row tile. K=64 = 2 x (16x16x32 f16 WMMA). N=64 = 4 tiles.
__global__ void __launch_bounds__(256)
gcn_gemm_wmma(const float* __restrict__ X, const _Float16* __restrict__ WhT,
              const float* __restrict__ dinv, float* __restrict__ g, float* __restrict__ s) {
    const int lane = threadIdx.x & 31;
    const int wave = threadIdx.x >> 5;
    const int tile = blockIdx.x * 8 + wave;
    if (tile >= NTILES) return;
    const int m0  = tile * 16;
    const int row = m0 + (lane & 15);
    const float* xrow = X + (size_t)row * D_C;

    // A fragments (contiguous k order, same as B):
    // lanes 0-15 hold row, K = k0+{0..15}; lanes 16-31: K = k0+{16..31}.
    v16h a[2];
#pragma unroll
    for (int ks = 0; ks < 2; ++ks) {
        const int base = ks * 32 + ((lane & 16) ? 16 : 0);
        v8f lo = *(const v8f*)(xrow + base);        // 32B aligned
        v8f hi = *(const v8f*)(xrow + base + 8);
#pragma unroll
        for (int i = 0; i < 8; ++i) { a[ks][i] = (_Float16)lo[i]; a[ks][8 + i] = (_Float16)hi[i]; }
    }

    const int col15 = lane & 15;
    const int koff  = (lane & 16) ? 16 : 0;   // B 32x16, contiguous k order: lanes 0-15 K=0..15, 16-31 K=16..31
    const int mbase = m0 + ((lane & 16) ? 8 : 0);

    // hoist dinv for the 8 rows this lane stores
    float dv[8];
#pragma unroll
    for (int r = 0; r < 8; ++r) dv[r] = dinv[mbase + r];

#pragma unroll
    for (int nt = 0; nt < 4; ++nt) {
        const int n = nt * 16 + col15;
        const _Float16* bp = WhT + (size_t)n * D_C;
        v16h b0 = *(const v16h*)(bp + koff);        // K = 0..31 slice, contiguous 32B
        v16h b1 = *(const v16h*)(bp + 32 + koff);   // K = 32..63 slice
        v8f acc = {};
        acc = __builtin_amdgcn_wmma_f32_16x16x32_f16(false, a[0], false, b0,
                                                     (short)0, acc, false, false);
        acc = __builtin_amdgcn_wmma_f32_16x16x32_f16(false, a[1], false, b1,
                                                     (short)0, acc, false, false);
        // D layout: lane -> column n; VGPR r -> row mbase+r
#pragma unroll
        for (int r = 0; r < 8; ++r) {
            const size_t idx = (size_t)(mbase + r) * D_C + n;
            const float v = acc[r] * dv[r];
            g[idx] = v;   // gather source (read-only during scatter)
            s[idx] = v;   // accumulator, pre-initialized with self-loop term
        }
    }
}

// ---------------- edge scatter: s[dst] += g[src], one wave32 per edge ----------------
__global__ void __launch_bounds__(256)
gcn_scatter(const float* __restrict__ g, float* __restrict__ s,
            const int* __restrict__ src, const int* __restrict__ dst, int nE) {
    const int wave = (int)((blockIdx.x * (size_t)blockDim.x + threadIdx.x) >> 5);
    const int lane = threadIdx.x & 31;
    if (wave >= nE) return;
    const int u = src[wave];              // uniform across the wave -> scalar loads
    const int v = dst[wave];
    const float* gs = g + (size_t)u * D_C;
    float*       sd = s + (size_t)v * D_C;
    unsafeAtomicAdd(&sd[lane],      gs[lane]);        // global_atomic_add_f32
    unsafeAtomicAdd(&sd[lane + 32], gs[lane + 32]);
}

// ---------------- finish: out = dinv[m]*s + b[n] (+relu); optional 2nd copy ----------------
__global__ void __launch_bounds__(256)
gcn_finish(const float* __restrict__ s, const float* __restrict__ dinv,
           const float* __restrict__ bias, float* __restrict__ out0,
           float* __restrict__ out1, int relu) {
    int i = blockIdx.x * blockDim.x + threadIdx.x;
    if (i >= N_NODES_C * D_C) return;
    const int m = i >> 6, n = i & 63;
    float v = dinv[m] * s[i] + bias[n];
    if (relu) v = fmaxf(v, 0.0f);
    out0[i] = v;
    if (out1) out1[i] = v;
}

// ---------------- launcher ----------------
extern "C" void kernel_launch(void* const* d_in, const int* in_sizes, int n_in,
                              void* d_out, int out_size, void* d_ws, size_t ws_size,
                              hipStream_t stream) {
    const float* x  = (const float*)d_in[0];
    const int*   ei = (const int*)  d_in[1];
    const float* W1 = (const float*)d_in[2];
    const float* b1 = (const float*)d_in[3];
    const float* W2 = (const float*)d_in[4];
    const float* b2 = (const float*)d_in[5];
    const int nE = in_sizes[1] / 2;
    const int* srcp = ei;
    const int* dstp = ei + nE;

    char* ws = (char*)d_ws;
    const size_t MB = 1u << 20;
    float*    deg  = (float*)(ws);               // 0.4 MB
    float*    dinv = (float*)(ws + 1 * MB);      // 0.4 MB
    _Float16* Wh1  = (_Float16*)(ws + 2 * MB);   // 8 KB
    _Float16* Wh2  = Wh1 + D_C * D_C;            // 8 KB
    float*    g    = (float*)(ws + 3  * MB);     // 25.6 MB
    float*    s    = (float*)(ws + 30 * MB);     // 25.6 MB
    float*    h1   = (float*)(ws + 57 * MB);     // 25.6 MB

    float* out = (float*)d_out;
    float* outA = out;                           // user||movie == h2 rows 0..N-1
    float* outB = out + (size_t)N_NODES_C * D_C; // full h2 again

    const int TB = 256;
    const int gN    = (N_NODES_C + TB - 1) / TB;
    const int gE    = (nE + TB - 1) / TB;
    const int gEw   = (nE + 7) / 8;                        // wave-per-edge
    const int gElem = (N_NODES_C * D_C + TB - 1) / TB;     // 25000
    const int gGemm = (NTILES + 7) / 8;                    // 8 waves/block

    // degree / normalization (shared by both layers)
    gcn_zero_f32<<<gN, TB, 0, stream>>>(deg, N_NODES_C);
    gcn_degree  <<<gE, TB, 0, stream>>>(dstp, deg, nE);
    gcn_dinv    <<<gN, TB, 0, stream>>>(deg, dinv, N_NODES_C);
    gcn_convW   <<<(2 * D_C * D_C + TB - 1) / TB, TB, 0, stream>>>(W1, W2, Wh1, Wh2);

    // layer 1
    gcn_gemm_wmma<<<gGemm, TB, 0, stream>>>(x, Wh1, dinv, g, s);
    gcn_scatter  <<<gEw,   TB, 0, stream>>>(g, s, srcp, dstp, nE);
    gcn_finish   <<<gElem, TB, 0, stream>>>(s, dinv, b1, h1, nullptr, 1);

    // layer 2
    gcn_gemm_wmma<<<gGemm, TB, 0, stream>>>(h1, Wh2, dinv, g, s);
    gcn_scatter  <<<gEw,   TB, 0, stream>>>(g, s, srcp, dstp, nE);
    gcn_finish   <<<gElem, TB, 0, stream>>>(s, dinv, b2, outA, outB, 0);
}
